// TransformerLayer_81956565942714
// MI455X (gfx1250) — matrix operand order repaired and checked by reading, not compiled, not run on hardware
//
#include <hip/hip_runtime.h>
#include <math.h>

// ---------------------------------------------------------------------------
// TransformerLayer forward for MI455X (gfx1250).
// bf16 WMMA GEMMs with double-buffered async global->LDS staging (ASYNCcnt)
// and software-pipelined LDS fragment loads (DScnt counted waits).
// ---------------------------------------------------------------------------

typedef __bf16 bf16;
typedef __attribute__((ext_vector_type(8)))  bf16  v8bf;
typedef __attribute__((ext_vector_type(16))) bf16  v16bf;
typedef __attribute__((ext_vector_type(8)))  float v8f;
typedef int v4i __attribute__((vector_size(16)));   // matches builtin's b128 payload

#define SEQ    4096
#define DMODEL 1024
#define DINT   512
#define DFF    4096

#define BM 128
#define BN 128
#define BK 32
#define LDS_STRIDE 40   // 32 data + 8 pad bf16; 80B rows keep 16B alignment

#if defined(__gfx1250__) && __has_builtin(__builtin_amdgcn_global_load_async_to_lds_b128)
#define HAVE_ASYNC 1
#define GLOBAL_AS __attribute__((address_space(1)))
#define LDS_AS    __attribute__((address_space(3)))
#else
#define HAVE_ASYNC 0
#endif

#if defined(__gfx1250__) && __has_builtin(__builtin_amdgcn_s_wait_asynccnt)
#define WAIT_ASYNC(n) __builtin_amdgcn_s_wait_asynccnt(n)
#elif defined(__gfx1250__)
#define WAIT_ASYNC(n) asm volatile("s_wait_asynccnt %0" :: "n"(n) : "memory")
#else
#define WAIT_ASYNC(n) do {} while (0)
#endif

__device__ __forceinline__ bf16 f2bf(float f) {
  unsigned u = __builtin_bit_cast(unsigned, f);
  u = (u + 0x7FFFu + ((u >> 16) & 1u)) >> 16;   // round-to-nearest-even
  return __builtin_bit_cast(bf16, (unsigned short)u);
}

// ---------------------------------------------------------------------------
// GEMM: D = epilogue( scale * A[M,K] @ B^T )  where
//   A  : bf16 [M,K] row-major, lda
//   B  : bf16 [N,K] row-major, ldb  (i.e. column j of the math-B is row j here)
// epilogue: (+bias[col]) (ReLU) (+resid[row,col] fp32). Outputs (each optional):
//   C   fp32  [M,N] (ldc)
//   CbN bf16  [M,N] (ldc)      -- normal-orientation bf16 copy
//   CbT bf16  [N,SEQ]          -- transposed bf16 copy (CbT[n*SEQ + m])
// ---------------------------------------------------------------------------
__global__ __launch_bounds__(256) void gemm_bf16_wmma(
    const bf16* __restrict__ A, int lda,
    const bf16* __restrict__ B, int ldb,
    const float* __restrict__ bias,
    const float* __restrict__ resid,
    float* __restrict__ C,
    bf16* __restrict__ CbN,
    bf16* __restrict__ CbT,
    int ldc, int Kdim, float scale, int do_bias, int do_relu, int do_resid)
{
  __shared__ __attribute__((aligned(16))) bf16 sA[2][BM * LDS_STRIDE];
  __shared__ __attribute__((aligned(16))) bf16 sB[2][BN * LDS_STRIDE];

  const int tid  = threadIdx.x;
  const int wave = tid >> 5;
  const int lane = tid & 31;
  const int hlf  = lane >> 4;
  const int l16  = lane & 15;

  const int row0 = blockIdx.y * BM;
  const int col0 = blockIdx.x * BN;

  v8f acc[8];
#pragma unroll
  for (int i = 0; i < 8; ++i) {
    v8f z = {0.f, 0.f, 0.f, 0.f, 0.f, 0.f, 0.f, 0.f};
    acc[i] = z;
  }

  // Stage one 128x32 bf16 tile of A and of B into LDS buffer `buf`.
  // 512 16-byte segments each; 256 threads x 2 ops; 4 async B128 per wave.
  auto stage = [&](int buf, int k0) {
#pragma unroll
    for (int op = 0; op < 2; ++op) {
      const int seg = op * 256 + tid;        // 0..511
      const int r   = seg >> 2;              // row/n 0..127
      const int s   = (seg & 3) << 3;        // k sub-offset 0,8,16,24 (bf16)
      const bf16* ga = A + (size_t)(row0 + r) * lda + k0 + s;
      const bf16* gb = B + (size_t)(col0 + r) * ldb + k0 + s;
      bf16* la = &sA[buf][r * LDS_STRIDE + s];
      bf16* lb = &sB[buf][r * LDS_STRIDE + s];
#if HAVE_ASYNC
      __builtin_amdgcn_global_load_async_to_lds_b128(
          (GLOBAL_AS v4i*)ga, (LDS_AS v4i*)la, 0, 0);
      __builtin_amdgcn_global_load_async_to_lds_b128(
          (GLOBAL_AS v4i*)gb, (LDS_AS v4i*)lb, 0, 0);
#else
      *(v8bf*)la = *(const v8bf*)ga;
      *(v8bf*)lb = *(const v8bf*)gb;
#endif
    }
  };

  const int nstages = Kdim / BK;
  stage(0, 0);

  for (int st = 0; st < nstages; ++st) {
    const int buf = st & 1;
    if (st + 1 < nstages) {
      stage(buf ^ 1, (st + 1) * BK);   // overlap next-tile DMA with compute
      WAIT_ASYNC(4);                   // only this stage's 4 ops must retire
    } else {
      WAIT_ASYNC(0);
    }
    __syncthreads();

    // A fragment (16x32): e0..7 = K 8*hlf+0..7, e8..15 = K 16+8*hlf+0..7
    const bf16* pa = &sA[buf][(wave * 16 + l16) * LDS_STRIDE];
    v8bf alo = *(const v8bf*)(pa + 8 * hlf);
    v8bf ahi = *(const v8bf*)(pa + 16 + 8 * hlf);
    v16bf afrag = __builtin_shufflevector(alo, ahi,
        0,1,2,3,4,5,6,7,8,9,10,11,12,13,14,15);

    // B fragments (32x16): e0..7 = K 8*hlf+0..7, e8..15 = K 16+8*hlf+0..7
    // Software-pipelined: loads for nb+1 are in flight before wmma(nb), so
    // the scheduler can use counted s_wait_dscnt instead of waiting to zero.
    const bf16* pb = &sB[buf][l16 * LDS_STRIDE + 8 * hlf];
    v8bf blo = *(const v8bf*)(pb);
    v8bf bhi = *(const v8bf*)(pb + 16);
#pragma unroll
    for (int nb = 0; nb < 8; ++nb) {
      v16bf bfrag = __builtin_shufflevector(blo, bhi,
          0,1,2,3,4,5,6,7,8,9,10,11,12,13,14,15);
      if (nb < 7) {
        const bf16* pbn = &sB[buf][((nb + 1) * 16 + l16) * LDS_STRIDE + 8 * hlf];
        blo = *(const v8bf*)(pbn);
        bhi = *(const v8bf*)(pbn + 16);
      }
      acc[nb] = __builtin_amdgcn_wmma_f32_16x16x32_bf16(
          false, afrag, false, bfrag, (short)0, acc[nb], false, false);
    }
    __syncthreads();
  }

  // Epilogue: C/D layout VGPR r -> row 8*hlf + r, col = lane&15
#pragma unroll
  for (int nb = 0; nb < 8; ++nb) {
    const int col = col0 + nb * 16 + l16;
    const float bb = do_bias ? bias[col] : 0.f;
#pragma unroll
    for (int r = 0; r < 8; ++r) {
      const int row = row0 + wave * 16 + hlf * 8 + r;
      float v = acc[nb][r] * scale + bb;
      if (do_relu) v = v > 0.f ? v : 0.f;
      const size_t off = (size_t)row * ldc + col;
      if (do_resid) v += resid[off];
      if (C)   C[off]   = v;
      if (CbN) CbN[off] = f2bf(v);
      if (CbT) CbT[(size_t)col * SEQ + row] = f2bf(v);
    }
  }
}

// Row softmax in place on fp32 p[rows,n]; also writes bf16 copy pb[rows,n].
__global__ __launch_bounds__(256) void softmax_rows(
    float* __restrict__ p, bf16* __restrict__ pb, int n)
{
  __shared__ float red[256];
  const int tid = threadIdx.x;
  float* row  = p  + (size_t)blockIdx.x * n;
  bf16*  rowb = pb + (size_t)blockIdx.x * n;

  float m = -3.402823466e38f;
  for (int i = tid; i < n; i += 256) m = fmaxf(m, row[i]);
  red[tid] = m; __syncthreads();
  for (int s = 128; s > 0; s >>= 1) {
    if (tid < s) red[tid] = fmaxf(red[tid], red[tid + s]);
    __syncthreads();
  }
  m = red[0]; __syncthreads();

  float sum = 0.f;
  for (int i = tid; i < n; i += 256) {
    float e = __expf(row[i] - m);
    row[i] = e;
    sum += e;
  }
  red[tid] = sum; __syncthreads();
  for (int s = 128; s > 0; s >>= 1) {
    if (tid < s) red[tid] += red[tid + s];
    __syncthreads();
  }
  const float inv = 1.f / red[0];
  for (int i = tid; i < n; i += 256) {
    float v = row[i] * inv;
    row[i]  = v;
    rowb[i] = f2bf(v);
  }
}

// fp32 -> bf16 elementwise (n multiple of 1024).
__global__ __launch_bounds__(256) void convert_f32_bf16(
    const float* __restrict__ in, bf16* __restrict__ out, int n)
{
  const int i4 = (blockIdx.x * 256 + threadIdx.x) * 4;
  if (i4 + 3 < n) {
    float4 v = *(const float4*)(in + i4);
    out[i4 + 0] = f2bf(v.x);
    out[i4 + 1] = f2bf(v.y);
    out[i4 + 2] = f2bf(v.z);
    out[i4 + 3] = f2bf(v.w);
  }
}

// W[K,N] fp32 -> Wt[N,K] bf16, 32x32 LDS tile transpose.
__global__ __launch_bounds__(256) void transpose_f32_bf16(
    const float* __restrict__ W, bf16* __restrict__ Wt, int K, int N)
{
  __shared__ float t[32][33];
  const int tx = threadIdx.x & 31;
  const int ty = threadIdx.x >> 5;      // 0..7
  const int n0 = blockIdx.x * 32;
  const int k0 = blockIdx.y * 32;
#pragma unroll
  for (int j = ty; j < 32; j += 8)
    t[j][tx] = W[(size_t)(k0 + j) * N + n0 + tx];
  __syncthreads();
#pragma unroll
  for (int j = ty; j < 32; j += 8)
    Wt[(size_t)(n0 + j) * K + k0 + tx] = f2bf(t[tx][j]);
}

extern "C" void kernel_launch(void* const* d_in, const int* in_sizes, int n_in,
                              void* d_out, int out_size, void* d_ws, size_t ws_size,
                              hipStream_t stream) {
  (void)in_sizes; (void)n_in; (void)out_size; (void)ws_size;

  const float* x  = (const float*)d_in[0];
  const float* Wq = (const float*)d_in[1];
  const float* bq = (const float*)d_in[2];
  const float* Wk = (const float*)d_in[3];
  const float* bk = (const float*)d_in[4];
  const float* Wv = (const float*)d_in[5];
  const float* bv = (const float*)d_in[6];
  const float* Wo = (const float*)d_in[7];
  const float* bo = (const float*)d_in[8];
  const float* W1 = (const float*)d_in[9];
  const float* b1 = (const float*)d_in[10];
  const float* W2 = (const float*)d_in[11];
  const float* b2 = (const float*)d_in[12];

  float* out  = (float*)d_out;                 // [SEQ, DMODEL]
  float* attn = out + (size_t)SEQ * DMODEL;    // [SEQ, SEQ]

  // ---- workspace layout ----
  float* res1 = (float*)d_ws;                              // fp32 [SEQ,DMODEL]
  bf16*  bp   = (bf16*)(res1 + (size_t)SEQ * DMODEL);
  bf16* xb    = bp; bp += (size_t)SEQ * DMODEL;            // x bf16
  bf16* Wqt   = bp; bp += (size_t)DINT * DMODEL;           // [512,1024]
  bf16* Wkt   = bp; bp += (size_t)DINT * DMODEL;
  bf16* Wvt   = bp; bp += (size_t)DINT * DMODEL;
  bf16* Wot   = bp; bp += (size_t)DMODEL * DINT;           // [1024,512]
  bf16* W1t   = bp; bp += (size_t)DFF * DMODEL;            // [4096,1024]
  bf16* W2t   = bp; bp += (size_t)DMODEL * DFF;            // [1024,4096]
  bf16* Qb    = bp; bp += (size_t)SEQ * DINT;              // [4096,512]
  bf16* Kmb   = bp; bp += (size_t)SEQ * DINT;              // [4096,512] == [N,K]
  bf16* Vtb   = bp; bp += (size_t)DINT * SEQ;              // [512,4096] transposed V
  bf16* O1b   = bp; bp += (size_t)SEQ * DINT;
  bf16* res1b = bp; bp += (size_t)SEQ * DMODEL;
  bf16* attnb = bp; bp += (size_t)SEQ * SEQ;
  bf16* hb    = bp;                                        // [SEQ,DFF]

  const dim3 blk(256);
  const float rs = 1.0f / sqrtf((float)DINT);

  // ---- one-time bf16 conversions / weight transposes ----
  convert_f32_bf16<<<dim3((SEQ * DMODEL) / 1024), blk, 0, stream>>>(x, xb, SEQ * DMODEL);
  transpose_f32_bf16<<<dim3(DINT / 32, DMODEL / 32), blk, 0, stream>>>(Wq, Wqt, DMODEL, DINT);
  transpose_f32_bf16<<<dim3(DINT / 32, DMODEL / 32), blk, 0, stream>>>(Wk, Wkt, DMODEL, DINT);
  transpose_f32_bf16<<<dim3(DINT / 32, DMODEL / 32), blk, 0, stream>>>(Wv, Wvt, DMODEL, DINT);
  transpose_f32_bf16<<<dim3(DMODEL / 32, DINT / 32), blk, 0, stream>>>(Wo, Wot, DINT, DMODEL);
  transpose_f32_bf16<<<dim3(DFF / 32, DMODEL / 32), blk, 0, stream>>>(W1, W1t, DMODEL, DFF);
  transpose_f32_bf16<<<dim3(DMODEL / 32, DFF / 32), blk, 0, stream>>>(W2, W2t, DFF, DMODEL);

  // ---- Q/K/V projections (bf16-only outputs; V also stored transposed) ----
  gemm_bf16_wmma<<<dim3(DINT / BN, SEQ / BM), blk, 0, stream>>>(
      xb, DMODEL, Wqt, DMODEL, bq, nullptr, nullptr, Qb, nullptr,
      DINT, DMODEL, 1.f, 1, 0, 0);
  gemm_bf16_wmma<<<dim3(DINT / BN, SEQ / BM), blk, 0, stream>>>(
      xb, DMODEL, Wkt, DMODEL, bk, nullptr, nullptr, Kmb, nullptr,
      DINT, DMODEL, 1.f, 1, 0, 0);
  gemm_bf16_wmma<<<dim3(DINT / BN, SEQ / BM), blk, 0, stream>>>(
      xb, DMODEL, Wvt, DMODEL, bv, nullptr, nullptr, nullptr, Vtb,
      DINT, DMODEL, 1.f, 1, 0, 0);

  // ---- scores = (Q @ K^T) / sqrt(d): fp32 straight into attn region ----
  gemm_bf16_wmma<<<dim3(SEQ / BN, SEQ / BM), blk, 0, stream>>>(
      Qb, DINT, Kmb, DINT, nullptr, nullptr, attn, nullptr, nullptr,
      SEQ, DINT, rs, 0, 0, 0);

  // ---- attn = softmax(scores) in place; bf16 copy for next GEMM ----
  softmax_rows<<<dim3(SEQ), blk, 0, stream>>>(attn, attnb, SEQ);

  // ---- O1 = attn @ V  (B = transposed V) ----
  gemm_bf16_wmma<<<dim3(DINT / BN, SEQ / BM), blk, 0, stream>>>(
      attnb, SEQ, Vtb, SEQ, nullptr, nullptr, nullptr, O1b, nullptr,
      DINT, SEQ, 1.f, 0, 0, 0);

  // ---- res1 = O1 @ Wo + bo + x ----
  gemm_bf16_wmma<<<dim3(DMODEL / BN, SEQ / BM), blk, 0, stream>>>(
      O1b, DINT, Wot, DINT, bo, x, res1, res1b, nullptr,
      DMODEL, DINT, 1.f, 1, 0, 1);

  // ---- h = relu(res1 @ W1 + b1) (bf16 only) ----
  gemm_bf16_wmma<<<dim3(DFF / BN, SEQ / BM), blk, 0, stream>>>(
      res1b, DMODEL, W1t, DMODEL, b1, nullptr, nullptr, hb, nullptr,
      DFF, DMODEL, 1.f, 1, 1, 0);

  // ---- out = h @ W2 + b2 + res1 ----
  gemm_bf16_wmma<<<dim3(DMODEL / BN, SEQ / BM), blk, 0, stream>>>(
      hb, DFF, W2t, DFF, b2, res1, out, nullptr, nullptr,
      DMODEL, DFF, 1.f, 1, 0, 1);
}
